// SlidingWindowAttention_14130442404432
// MI455X (gfx1250) — matrix operand order repaired and checked by reading, not compiled, not run on hardware
//
#include <hip/hip_runtime.h>
#include <hip/hip_bf16.h>

typedef _Float16 h16 __attribute__((ext_vector_type(16)));
typedef _Float16 h8  __attribute__((ext_vector_type(8)));
typedef float    f8  __attribute__((ext_vector_type(8)));

#define WMMA_F16(a, b, c) \
  __builtin_amdgcn_wmma_f32_16x16x32_f16(false, (a), false, (b), (short)0, (c), false, false)

static constexpr int Lq = 8192;
static constexpr int Dm = 1024;
static constexpr int Hh = 16;
static constexpr int HD = 64;
static constexpr int Wn = 256;   // window / chunk size

// LDS byte offset of a generic pointer into __shared__ space (AS3 ptrs are 32-bit)
__device__ inline unsigned lds_off(const void* p) {
  return (unsigned)(unsigned long long)(__attribute__((address_space(3))) const void*)p;
}
// CDNA5 async global->LDS copy, 16 bytes per lane; ASYNCcnt-tracked.
__device__ inline void async_g2l_b128(unsigned lds_byte_off, const void* gaddr) {
  asm volatile("global_load_async_to_lds_b128 %0, %1, off"
               :: "v"(lds_byte_off), "v"((unsigned long long)gaddr) : "memory");
}
__device__ inline void wait_async0() {
  asm volatile("s_wait_asynccnt 0" ::: "memory");
}

// ---------------------------------------------------------------- casts
__global__ void cast_f32_f16(const float* __restrict__ in, _Float16* __restrict__ out, int n) {
  int i = blockIdx.x * blockDim.x + threadIdx.x;
  if (i < n) out[i] = (_Float16)in[i];
}

// in[rows][cols] (row-major) -> out[cols][rows]  (transposed, f16)
__global__ void transpose_cast(const float* __restrict__ in, _Float16* __restrict__ out,
                               int rows, int cols) {
  int i = blockIdx.x * blockDim.x + threadIdx.x;
  if (i < rows * cols) {
    int r = i / cols, c = i % cols;
    out[(size_t)c * rows + r] = (_Float16)in[i];
  }
}

// ---------------------------------------------------------------- GEMM
// C[M,N] = A[M,K] * B[K,N], with B supplied transposed as Bt[N,K].
// Block = 256 threads = 8 waves (4 M-waves x 2 N-waves); block tile = 128x128.
// Wave tile = 32x64 (8 WMMAs per 32-deep k-step).
// B tile (32k x 128n) is double-buffered in LDS via async global->LDS loads.
template <bool STORE_HALF>
__global__ void gemm_wmma(const _Float16* __restrict__ A, const _Float16* __restrict__ Bt,
                          void* __restrict__ Cout, int M, int N, int K) {
  __shared__ __attribute__((aligned(128))) _Float16 Bs[2][128][32];

  const int lane = threadIdx.x & 31;
  const int wave = threadIdx.x >> 5;
  const int hi = lane >> 4, lo = lane & 15;
  const int waveM = wave & 3, waveN = wave >> 2;
  const int mbase = blockIdx.y * 128 + waveM * 32;
  const int nbase = blockIdx.x * 128 + waveN * 64;

  // Cooperative B staging: thread t handles row n = t/2, k-halves [(t&1)*16 .. +15]
  const int sn = threadIdx.x >> 1;
  const int sk = (threadIdx.x & 1) * 16;
  const _Float16* gstage = Bt + (size_t)(blockIdx.x * 128 + sn) * K + sk;

  auto stage = [&](int buf, int kk) {
    const _Float16* g = gstage + kk;
    unsigned l = lds_off(&Bs[buf][sn][sk]);
    async_g2l_b128(l, g);                       // first 8 halves
    asm volatile("global_load_async_to_lds_b128 %0, %1, off offset:16"
                 :: "v"(l), "v"((unsigned long long)g) : "memory");  // next 8 halves
  };

  f8 acc[2][4] = {};

  stage(0, 0);
  wait_async0();
  __syncthreads();

  const int nk = K / 32;
  int buf = 0;
  for (int t = 0; t < nk; ++t) {
    if (t + 1 < nk) stage(buf ^ 1, (t + 1) * 32);  // async prefetch next tile

    const int kk = t * 32;
    // A fragments: 16x32, lane=M-row; element j -> K = hi*16 + j (contiguous)
    h16 a[2];
#pragma unroll
    for (int rt = 0; rt < 2; ++rt) {
      const _Float16* pa = A + (size_t)(mbase + rt * 16 + lo) * K + kk + hi * 16;
      h8 alo = *(const h8*)pa;
      h8 ahi = *(const h8*)(pa + 8);
#pragma unroll
      for (int j = 0; j < 8; ++j) { a[rt][j] = alo[j]; a[rt][8 + j] = ahi[j]; }
    }
#pragma unroll
    for (int nt = 0; nt < 4; ++nt) {
      // B fragment from LDS: lane=N-col; element j -> K = hi*16 + j (contiguous)
      const int nloc = waveN * 64 + nt * 16 + lo;
      h16 b = *(const h16*)(&Bs[buf][nloc][hi * 16]);
      acc[0][nt] = WMMA_F16(a[0], b, acc[0][nt]);
      acc[1][nt] = WMMA_F16(a[1], b, acc[1][nt]);
    }

    wait_async0();      // next tile's asyncs overlapped with this tile's WMMAs
    __syncthreads();
    buf ^= 1;
  }

#pragma unroll
  for (int rt = 0; rt < 2; ++rt)
#pragma unroll
    for (int nt = 0; nt < 4; ++nt) {
      const int ncol = nbase + nt * 16 + lo;
#pragma unroll
      for (int r = 0; r < 8; ++r) {
        const int m = mbase + rt * 16 + hi * 8 + r;  // C: elem r -> M = hi*8+r, N = lane
        if constexpr (STORE_HALF)
          ((_Float16*)Cout)[(size_t)m * N + ncol] = (_Float16)acc[rt][nt][r];
        else
          ((float*)Cout)[(size_t)m * N + ncol] = acc[rt][nt][r];
      }
    }
}

// ---------------------------------------------------------------- band attention
// qh  : [L, D] f16   (col = h*64 + d)
// kvh : [L, 128] f16 (cols 0..63 = K head, 64..127 = V head; shared across H)
// o16 : [L, D] f16
// Grid: (2*nb, H). Block = 256 threads = 8 waves; wave handles one 16-query tile.
__global__ void band_attn(const _Float16* __restrict__ qh, const _Float16* __restrict__ kvh,
                          _Float16* __restrict__ o16) {
  __shared__ __attribute__((aligned(64))) _Float16 Vt[64][32];       // transposed V block
  __shared__ __attribute__((aligned(64))) _Float16 Pst[8][16][32];   // per-wave P staging

  const int lane = threadIdx.x & 31;
  const int wave = threadIdx.x >> 5;
  const int hi = lane >> 4, lo = lane & 15;
  const int h = blockIdx.y;
  const int chunk = blockIdx.x >> 1;
  const int half  = blockIdx.x & 1;
  const int qrow0 = chunk * Wn + (half * 8 + wave) * 16;
  const int qrow  = qrow0 + lo;

  // Q fragments for d=0..31 and d=32..63 (loop invariant)
  const _Float16* pq = qh + (size_t)qrow * Dm + h * HD;
  h16 a0, a1;
  {
    h8 t0 = *(const h8*)(pq + hi * 16);
    h8 t1 = *(const h8*)(pq + 8 + hi * 16);
    h8 t2 = *(const h8*)(pq + 32 + hi * 16);
    h8 t3 = *(const h8*)(pq + 40 + hi * 16);
#pragma unroll
    for (int j = 0; j < 8; ++j) {
      a0[j] = t0[j]; a0[8 + j] = t1[j];
      a1[j] = t2[j]; a1[8 + j] = t3[j];
    }
  }

  float rmax[8], rsum[8];
#pragma unroll
  for (int r = 0; r < 8; ++r) { rmax[r] = -1e30f; rsum[r] = 0.f; }
  f8 acc[4] = {};

  const int kwin0 = chunk * Wn - Wn;  // window start (may be negative; masked)

  for (int kb = 0; kb < 24; ++kb) {   // 24 blocks of 32 keys = 768-key window
    const int kblk = kwin0 + kb * 32;

    __syncthreads();
    {   // cooperative transposed V staging: 32 keys x 64 dims
      const int keyl = threadIdx.x >> 3;          // 0..31
      const int dg = (threadIdx.x & 7) * 8;       // 0,8,...,56
      int krow = kblk + keyl;
      krow = krow < 0 ? 0 : (krow >= Lq ? Lq - 1 : krow);
      h8 v = *(const h8*)(kvh + (size_t)krow * 128 + 64 + dg);
#pragma unroll
      for (int j = 0; j < 8; ++j) Vt[dg + j][keyl] = v[j];
    }
    __syncthreads();

    // S = Q K^T for two 16-key tiles (contract d=64 as 2x WMMA)
    f8 s[2];
#pragma unroll
    for (int st = 0; st < 2; ++st) {
      int krow = kblk + st * 16 + lo;
      int krc = krow < 0 ? 0 : (krow >= Lq ? Lq - 1 : krow);
      const _Float16* pk = kvh + (size_t)krc * 128;
      h16 bk0 = *(const h16*)(pk + hi * 16);        // d = hi*16 + j  (0..31)
      h16 bk1 = *(const h16*)(pk + 32 + hi * 16);   // d = 32 + hi*16 + j
      f8 c = {};
      c = WMMA_F16(a0, bk0, c);
      c = WMMA_F16(a1, bk1, c);
      s[st] = c;
    }

    // mask + block row-max
    float bm[8];
#pragma unroll
    for (int r = 0; r < 8; ++r) bm[r] = -1e30f;
#pragma unroll
    for (int st = 0; st < 2; ++st) {
      const int kcol = kblk + st * 16 + lo;
#pragma unroll
      for (int r = 0; r < 8; ++r) {
        const int qi = qrow0 + hi * 8 + r;
        const int rel = kcol - qi;
        const bool ok = (rel >= -Wn) && (rel <= Wn) && (kcol >= 0) && (kcol < Lq);
        float v = ok ? s[st][r] : -1e30f;
        s[st][r] = v;
        bm[r] = fmaxf(bm[r], v);
      }
    }
#pragma unroll
    for (int r = 0; r < 8; ++r) {   // width-16 reduction matches C-layout lane groups
      float m = bm[r];
      m = fmaxf(m, __shfl_xor(m, 1, 16));
      m = fmaxf(m, __shfl_xor(m, 2, 16));
      m = fmaxf(m, __shfl_xor(m, 4, 16));
      m = fmaxf(m, __shfl_xor(m, 8, 16));
      bm[r] = m;
    }

    float scale[8], bs[8];
#pragma unroll
    for (int r = 0; r < 8; ++r) {
      float nm = fmaxf(rmax[r], bm[r]);
      scale[r] = __expf(rmax[r] - nm);
      rmax[r] = nm;
      bs[r] = 0.f;
    }

    // P = exp(S - m), stage to per-wave LDS in C layout
#pragma unroll
    for (int st = 0; st < 2; ++st) {
#pragma unroll
      for (int r = 0; r < 8; ++r) {
        float v = s[st][r];
        float p = (v > -1e29f) ? __expf(v - rmax[r]) : 0.f;
        bs[r] += p;
        Pst[wave][hi * 8 + r][st * 16 + lo] = (_Float16)p;
      }
    }
#pragma unroll
    for (int r = 0; r < 8; ++r) {
      float sv = bs[r];
      sv += __shfl_xor(sv, 1, 16);
      sv += __shfl_xor(sv, 2, 16);
      sv += __shfl_xor(sv, 4, 16);
      sv += __shfl_xor(sv, 8, 16);
      rsum[r] = rsum[r] * scale[r] + sv;
    }
#pragma unroll
    for (int dt = 0; dt < 4; ++dt)
#pragma unroll
      for (int r = 0; r < 8; ++r) acc[dt][r] *= scale[r];

    // reload P as a 16x32 A fragment
    h16 pa;
    {
      const _Float16* pp = &Pst[wave][lo][0];
      h8 plo = *(const h8*)(pp + hi * 16);
      h8 phi2 = *(const h8*)(pp + 8 + hi * 16);
#pragma unroll
      for (int j = 0; j < 8; ++j) { pa[j] = plo[j]; pa[8 + j] = phi2[j]; }
    }

    // O += P @ V : B fragment from transposed LDS tile (contiguous ds loads)
#pragma unroll
    for (int dt = 0; dt < 4; ++dt) {
      h16 bv = *(const h16*)(&Vt[dt * 16 + lo][0] + hi * 16);
      acc[dt] = WMMA_F16(pa, bv, acc[dt]);
    }
  }

  // normalize and store f16
#pragma unroll
  for (int r = 0; r < 8; ++r) {
    const float inv = 1.f / rsum[r];
    const int m = qrow0 + hi * 8 + r;
#pragma unroll
    for (int dt = 0; dt < 4; ++dt) {
      const int col = h * HD + dt * 16 + lo;
      o16[(size_t)m * Dm + col] = (_Float16)(acc[dt][r] * inv);
    }
  }
}

// ---------------------------------------------------------------- launch
extern "C" void kernel_launch(void* const* d_in, const int* in_sizes, int n_in,
                              void* d_out, int out_size, void* d_ws, size_t ws_size,
                              hipStream_t stream) {
  (void)in_sizes; (void)n_in; (void)out_size; (void)ws_size;
  const float* q    = (const float*)d_in[0];
  const float* kv   = (const float*)d_in[1];
  const float* w_q  = (const float*)d_in[2];
  const float* w_kv = (const float*)d_in[3];
  const float* w_o  = (const float*)d_in[4];
  float* out = (float*)d_out;

  char* ws = (char*)d_ws;
  size_t off = 0;
  auto alloc = [&](size_t bytes) { char* p = ws + off; off += (bytes + 255) & ~size_t(255); return p; };
  _Float16* q16   = (_Float16*)alloc((size_t)Lq * Dm * 2);
  _Float16* kv16  = (_Float16*)alloc((size_t)Lq * Dm * 2);
  _Float16* wqT   = (_Float16*)alloc((size_t)Dm * Dm * 2);
  _Float16* wkvT  = (_Float16*)alloc((size_t)Dm * 2 * HD * 2);
  _Float16* woT   = (_Float16*)alloc((size_t)Dm * Dm * 2);
  _Float16* qh16  = (_Float16*)alloc((size_t)Lq * Dm * 2);
  _Float16* kvh16 = (_Float16*)alloc((size_t)Lq * 2 * HD * 2);
  _Float16* o16   = (_Float16*)alloc((size_t)Lq * Dm * 2);

  const int nQ = Lq * Dm;
  cast_f32_f16<<<(nQ + 255) / 256, 256, 0, stream>>>(q, q16, nQ);
  cast_f32_f16<<<(nQ + 255) / 256, 256, 0, stream>>>(kv, kv16, nQ);
  transpose_cast<<<(Dm * Dm + 255) / 256, 256, 0, stream>>>(w_q, wqT, Dm, Dm);
  transpose_cast<<<(Dm * 2 * HD + 255) / 256, 256, 0, stream>>>(w_kv, wkvT, Dm, 2 * HD);
  transpose_cast<<<(Dm * Dm + 255) / 256, 256, 0, stream>>>(w_o, woT, Dm, Dm);

  // qh = q @ w_q   [8192,1024]
  gemm_wmma<true><<<dim3(Dm / 128, Lq / 128), 256, 0, stream>>>(q16, wqT, qh16, Lq, Dm, Dm);
  // kvh = kv @ w_kv  [8192,128]
  gemm_wmma<true><<<dim3(1, Lq / 128), 256, 0, stream>>>(kv16, wkvT, kvh16, Lq, 2 * HD, Dm);
  // sliding-window attention -> o16 [8192,1024]
  band_attn<<<dim3(2 * (Lq / Wn), Hh), 256, 0, stream>>>(qh16, kvh16, o16);
  // out = o @ w_o  (f32 store)
  gemm_wmma<false><<<dim3(Dm / 128, Lq / 128), 256, 0, stream>>>(o16, woT, out, Lq, Dm, Dm);
}
